// LinearNetwork_44607530336375
// MI455X (gfx1250) — matrix operand order repaired and checked by reading, not compiled, not run on hardware
//
#include <hip/hip_runtime.h>

typedef __attribute__((ext_vector_type(16))) _Float16 v16h;
typedef __attribute__((ext_vector_type(8)))  float    v8f;

#define FDIM  63    // 3 + 3*2*10
#define KPAD  64    // pad K to 64 (enc[63]=0, W col63=0)
#define NROWS 384   // 3 * 128 clusters

// One wave32 processes 16 points/tile:
//   A[16x64] = f16 positional encodings (WMMA A layout)
//   B[64x48] = gathered W rows for the 48 outputs this tile needs
//   D[16x48] via 6x v_wmma_f32_16x16x32_f16; each lane stores its diagonal value.
__global__ __launch_bounds__(256) void nerf_rgb_wmma_kernel(
    const float* __restrict__ X, const int* __restrict__ cid,
    const float* __restrict__ W, float* __restrict__ out,
    int N, int nTiles, int nWaves)
{
  __shared__ __align__(128) _Float16 Wl[NROWS * KPAD];  // 48 KB

  // ---- stage W into LDS as f16, zero-padded column 63 ----
  for (int i = threadIdx.x; i < NROWS * KPAD; i += blockDim.x) {
    int r = i >> 6, k = i & 63;
    float w = (k < FDIM) ? W[r * FDIM + k] : 0.0f;
    Wl[i] = (_Float16)w;
  }
  __syncthreads();

  const int lane = threadIdx.x & 31;
  const int n    = lane & 15;   // point-in-tile (A rows) / column-in-tile (B cols)
  const int h    = lane >> 4;   // half-wave: selects K-slice per WMMA layout
  const int wave = blockIdx.x * (blockDim.x >> 5) + (threadIdx.x >> 5);

  for (int tile = wave; tile < nTiles; tile += nWaves) {
    const int base = tile * 16;

    // ---- load this lane's point ----
    int p = base + n; if (p >= N) p = N - 1;
    const float x0 = X[p * 3 + 0];
    const float x1 = X[p * 3 + 1];
    const float x2 = X[p * 3 + 2];

    // ---- positional encoding: enc[0..2]=xyz, [3..32]=sin, [33..62]=cos, [63]=0
    // anchors at f=0 and f=5 (__sincosf) + double-angle recurrences
    float enc[KPAD];
    enc[0] = x0; enc[1] = x1; enc[2] = x2; enc[63] = 0.0f;
    const float xs[3] = {x0, x1, x2};
    #pragma unroll
    for (int d = 0; d < 3; ++d) {
      float s, c;
      __sincosf(xs[d], &s, &c);
      enc[3 + 10*d + 0] = s; enc[33 + 10*d + 0] = c;
      #pragma unroll
      for (int f = 1; f < 5; ++f) {
        float s2 = 2.0f * s * c;
        float c2 = fmaf(-2.0f * s, s, 1.0f);   // cos2t = 1 - 2 sin^2 t
        s = s2; c = c2;
        enc[3 + 10*d + f] = s; enc[33 + 10*d + f] = c;
      }
      __sincosf(xs[d] * 32.0f, &s, &c);
      enc[3 + 10*d + 5] = s; enc[33 + 10*d + 5] = c;
      #pragma unroll
      for (int f = 6; f < 10; ++f) {
        float s2 = 2.0f * s * c;
        float c2 = fmaf(-2.0f * s, s, 1.0f);
        s = s2; c = c2;
        enc[3 + 10*d + f] = s; enc[33 + 10*d + f] = c;
      }
    }

    // ---- pack A for the 16x16x32 f16 WMMA in contiguous k order (2 K-chunks) ----
    // VGPR v: K = 32c + 16h + 2v (+1)
    v16h a[2];
    #pragma unroll
    for (int c = 0; c < 2; ++c) {
      #pragma unroll
      for (int v = 0; v < 8; ++v) {
        const int KA = 32 * c + 2 * v;
        float lo = h ? enc[KA + 16] : enc[KA];
        float hi = h ? enc[KA + 17] : enc[KA + 1];
        a[c][2 * v]     = (_Float16)lo;
        a[c][2 * v + 1] = (_Float16)hi;
      }
    }

    // ---- 3 column tiles: gather B rows, WMMA, extract diagonal, store ----
    #pragma unroll
    for (int t = 0; t < 3; ++t) {
      const int j  = 16 * t + n;      // global output column 0..47
      const int m  = j / 3;           // which point in the tile owns column j
      const int ch = j - 3 * m;       // rgb channel
      int q = base + m; if (q >= N) q = N - 1;
      const int row = 3 * cid[q] + ch;

      // B in contiguous k order matching A: lanes0-15 K=0..15, lanes16-31 K=16..31 (per chunk)
      const int boff = row * KPAD + 16 * h;   // halves; 32B-aligned
      v16h b0 = *reinterpret_cast<const v16h*>(&Wl[boff]);
      v16h b1 = *reinterpret_cast<const v16h*>(&Wl[boff + 32]);

      v8f acc = {};
      acc = __builtin_amdgcn_wmma_f32_16x16x32_f16(false, a[0], false, b0,
                                                   (short)0, acc, false, false);
      acc = __builtin_amdgcn_wmma_f32_16x16x32_f16(false, a[1], false, b1,
                                                   (short)0, acc, false, false);

      // C layout: VGPR i holds M = i + 8*h, N = lane&15. We want D[m][j].
      float val = acc[0];
      #pragma unroll
      for (int i = 1; i < 8; ++i)
        if ((m & 7) == i) val = acc[i];

      const long long o = (long long)tile * 48 + j;  // == 3*(base+m)+ch
      if (((m >> 3) == h) && (o < 3LL * N))
        out[o] = val;
    }
  }
}

extern "C" void kernel_launch(void* const* d_in, const int* in_sizes, int n_in,
                              void* d_out, int out_size, void* d_ws, size_t ws_size,
                              hipStream_t stream) {
  const float* X   = (const float*)d_in[0];
  const int*   cid = (const int*)  d_in[1];
  const float* W   = (const float*)d_in[2];
  float*       out = (float*)d_out;

  const int N      = in_sizes[0] / 3;         // 1048576
  const int nTiles = (N + 15) / 16;           // 65536
  const int wavesPerBlock = 8;                // 256 threads, wave32
  int blocks = (nTiles + wavesPerBlock - 1) / wavesPerBlock;
  if (blocks > 2048) blocks = 2048;           // waves loop over tiles
  const int nWaves = blocks * wavesPerBlock;

  hipLaunchKernelGGL(nerf_rgb_wmma_kernel, dim3(blocks), dim3(256), 0, stream,
                     X, cid, W, out, N, nTiles, nWaves);
}
